// UltrasoundWNO1d_1288490189220
// MI455X (gfx1250) — matrix operand order repaired and checked by reading, not compiled, not run on hardware
//
#include <hip/hip_runtime.h>
#include <math.h>

typedef __attribute__((ext_vector_type(16))) _Float16 v16h;
typedef __attribute__((ext_vector_type(8)))  float    v8f;

__constant__ float c_dec_lo[16] = {
  -0.0033824159510061256f, -0.0005421323317911481f, 0.03169508781149298f,  0.007607487324917605f,
  -0.1432942383508097f,    -0.061273359067658524f,  0.4813596512583722f,   0.7771857517005235f,
   0.3644418948353314f,    -0.05194583810770904f,  -0.027219029917056003f, 0.049137179673607506f,
   0.003808752013890615f,  -0.01495225833704823f,  -0.0003029205147213668f,0.0018899503327594609f };

__device__ __forceinline__ float gelu_f(float x){
  return 0.5f * x * (1.0f + erff(x * 0.7071067811865475f));
}

// -------- lift --------
__global__ __launch_bounds__(256) void lift_kernel(
    const float* __restrict__ x, const float* __restrict__ lw,
    const float* __restrict__ lb, float* __restrict__ h,
    int B, int C, int S){
  size_t idx = (size_t)blockIdx.x * blockDim.x + threadIdx.x;
  size_t total = (size_t)B * C * S;
  if (idx >= total) return;
  int l = (int)(idx % S); size_t bc = idx / S; int c = (int)(bc % C); int b = (int)(bc / C);
  float g = -1.0f + 2.0f * (float)l / (float)(S - 1);
  h[idx] = lw[2*c] * x[(size_t)b*S + l] + lw[2*c+1] * g + lb[c];
}

// -------- instance norm over length axis --------
__global__ __launch_bounds__(256) void inorm_kernel(
    const float* __restrict__ x, float* __restrict__ y, int S){
  int row = blockIdx.x;
  const float* xr = x + (size_t)row * S;
  float* yr = y + (size_t)row * S;
  float s = 0.f, s2 = 0.f;
  for (int i = threadIdx.x; i < S; i += blockDim.x){ float v = xr[i]; s += v; s2 += v*v; }
  __shared__ float sh[256], sh2[256];
  sh[threadIdx.x] = s; sh2[threadIdx.x] = s2; __syncthreads();
  for (int st = blockDim.x >> 1; st > 0; st >>= 1){
    if ((int)threadIdx.x < st){ sh[threadIdx.x] += sh[threadIdx.x+st]; sh2[threadIdx.x] += sh2[threadIdx.x+st]; }
    __syncthreads();
  }
  float mean = sh[0] / (float)S;
  float var  = sh2[0] / (float)S - mean*mean;
  float inv  = rsqrtf(var + 1e-5f);
  for (int i = threadIdx.x; i < S; i += blockDim.x) yr[i] = (xr[i] - mean) * inv;
}

// -------- DWT (sym pad 14/15, stride 2, 16-tap) --------
__global__ __launch_bounds__(256) void dwt_kernel(
    const float* __restrict__ x, float* __restrict__ a, float* __restrict__ d,
    int Lin, int Lout){
  int row = blockIdx.y;
  int i = blockIdx.x * blockDim.x + threadIdx.x;
  if (i >= Lout) return;
  const float* xr = x + (size_t)row * Lin;
  float alo = 0.f, ahi = 0.f;
  #pragma unroll
  for (int t = 0; t < 16; ++t){
    int j = 2*i + t - 14;
    if (j < 0)    j = -j - 1;
    if (j >= Lin) j = 2*Lin - 1 - j;
    float xv = xr[j];
    alo += xv * c_dec_lo[15 - t];
    ahi += ((t & 1) ? -xv : xv) * c_dec_lo[t];
  }
  a[(size_t)row * Lout + i] = alo;
  d[(size_t)row * Lout + i] = ahi;
}

// -------- IDWT (lhs_dilation 2, pad (1,1)); Lout = 2*La - 14 --------
__global__ __launch_bounds__(256) void idwt_kernel(
    const float* __restrict__ a, const float* __restrict__ d, float* __restrict__ y,
    int La, int Lout, int stride_a, int stride_d){
  int row = blockIdx.y;
  int i = blockIdx.x * blockDim.x + threadIdx.x;
  if (i >= Lout) return;
  const float* ar = a + (size_t)row * stride_a;
  const float* dr = d + (size_t)row * stride_d;
  float acc = 0.f;
  int t0 = ((i & 1) == 0) ? 1 : 0;
  #pragma unroll
  for (int t = t0; t < 16; t += 2){
    int m = (i + t - 1) >> 1;
    if (m >= 0 && m < La){
      float rlo = c_dec_lo[t];
      float rhi = ((t & 1) ? 1.0f : -1.0f) * c_dec_lo[15 - t];
      acc += ar[m] * rlo + dr[m] * rhi;
    }
  }
  y[(size_t)row * Lout + i] = acc;
}

// -------- pack/transpose: X (B,K,N) f32 -> XT (B,N,K) f16 via LDS 32x32 tile ----
__global__ __launch_bounds__(256) void packT_kernel(
    const float* __restrict__ X, _Float16* __restrict__ XT, int K, int N){
  __shared__ _Float16 t[32][33];
  int n0 = blockIdx.x * 32, k0 = blockIdx.y * 32;
  size_t xb = (size_t)blockIdx.z * K * N;
  size_t tb = (size_t)blockIdx.z * N * K;
  int tx = threadIdx.x & 31, ty = threadIdx.x >> 5;   // 32 x 8
  int n = n0 + tx; int nn = (n < N) ? n : (N - 1);
  #pragma unroll
  for (int kk = ty; kk < 32; kk += 8)
    t[kk][tx] = (_Float16)X[xb + (size_t)(k0 + kk) * N + nn];
  __syncthreads();
  #pragma unroll
  for (int r = ty; r < 32; r += 8){
    int nw = n0 + r;
    if (nw < N) XT[tb + (size_t)nw * K + k0 + tx] = t[tx][r];
  }
}

// ======== WMMA GEMM on packed f16 input: Y[b,m,n] = act(W X + bias) (+Res) ========
// XT is (B, N, K) f16: one lane's B-fragment = 32 contiguous bytes -> 2x b128 loads.
// KT > 0: compile-time K (fully unrolled). KT == 0: runtime K fallback.
template<int KT>
__global__ __launch_bounds__(128) void gemm_wmma_kernel(
    const float* __restrict__ W, const float* __restrict__ bias,
    const _Float16* __restrict__ XT, const float* __restrict__ Res,
    float* __restrict__ Y, int M, int Kr, int N, int do_gelu){
  const int K = (KT > 0) ? KT : Kr;
  int lane = threadIdx.x & 31;
  int wv   = threadIdx.x >> 5;
  int l15  = lane & 15;
  int hi   = (lane >> 4) & 1;
  int n0 = (blockIdx.x * 8 + wv * 2) * 16;   // 4 waves x 2 tiles = 128 cols/block
  int m0 = blockIdx.y * 16;
  size_t tb = (size_t)blockIdx.z * N * K;
  size_t yb = (size_t)blockIdx.z * M * N;
  int na = n0 + l15;
  int nb = na + 16;
  int nlda = (na < N) ? na : (N - 1);        // clamp: OOB cols never stored
  int nldb = (nb < N) ? nb : (N - 1);
  const _Float16* xpa = XT + tb + (size_t)nlda * K + (hi ? 16 : 0);
  const _Float16* xpb = XT + tb + (size_t)nldb * K + (hi ? 16 : 0);
  const float*    wp  = W + (size_t)(m0 + l15) * K + (hi ? 16 : 0);
  v8f acc0 = {}, acc1 = {};
  #pragma unroll
  for (int kb = 0; kb < K; kb += 32){
    v16h af;
    #pragma unroll
    for (int idx = 0; idx < 16; ++idx)
      af[idx] = (_Float16)wp[kb + idx];   // A 16x32 f16 layout
    v16h bf0 = *(const v16h*)(xpa + kb);                         // B 32x16 f16 layout
    v16h bf1 = *(const v16h*)(xpb + kb);
    acc0 = __builtin_amdgcn_wmma_f32_16x16x32_f16(false, af, false, bf0, (short)0, acc0, false, false);
    acc1 = __builtin_amdgcn_wmma_f32_16x16x32_f16(false, af, false, bf1, (short)0, acc1, false, false);
  }
  #pragma unroll
  for (int tile = 0; tile < 2; ++tile){
    int n = (tile == 0) ? na : nb;
    v8f acc = (tile == 0) ? acc0 : acc1;
    if (n < N){
      #pragma unroll
      for (int r = 0; r < 8; ++r){
        int m = m0 + r + (hi ? 8 : 0);
        float v = acc[r] + (bias ? bias[m] : 0.0f);
        if (do_gelu) v = gelu_f(v);
        size_t o = yb + (size_t)m * N + n;
        if (Res) v += Res[o];
        Y[o] = v;
      }
    }
  }
}

// -------- k=3 channel conv on packed f16 input, true zero pad via f16 mask ------
template<int KT>
__global__ __launch_bounds__(128) void conv3_wmma_kernel(
    const float* __restrict__ W, const float* __restrict__ bias,
    const _Float16* __restrict__ XT, float* __restrict__ Y, int M, int Kr, int N){
  const int K = (KT > 0) ? KT : Kr;
  int lane = threadIdx.x & 31;
  int wv   = threadIdx.x >> 5;
  int l15  = lane & 15;
  int hi   = (lane >> 4) & 1;
  int n0 = (blockIdx.x * 4 + wv) * 16;
  int m0 = blockIdx.y * 16;
  size_t tb = (size_t)blockIdx.z * N * K;
  size_t yb = (size_t)blockIdx.z * M * N;
  const float* wbase = W + (size_t)(m0 + l15) * K * 3;
  v8f acc = {};
  #pragma unroll
  for (int sh = 0; sh < 3; ++sh){
    int n = n0 + l15 + sh - 1;
    int nld = (n < 0) ? 0 : ((n < N) ? n : (N - 1));
    _Float16 hm = (_Float16)((n >= 0 && n < N) ? 1.0f : 0.0f);
    const _Float16* xp = XT + tb + (size_t)nld * K + (hi ? 16 : 0);
    #pragma unroll
    for (int kb = 0; kb < K; kb += 32){
      v16h af;
      #pragma unroll
      for (int idx = 0; idx < 16; ++idx){
        int k = kb + idx + (hi ? 16 : 0);
        af[idx] = (_Float16)wbase[k * 3 + sh];
      }
      v16h bf = *(const v16h*)(xp + kb);
      #pragma unroll
      for (int idx = 0; idx < 16; ++idx) bf[idx] = bf[idx] * hm;   // v_pk_mul_f16
      acc = __builtin_amdgcn_wmma_f32_16x16x32_f16(false, af, false, bf, (short)0, acc, false, false);
    }
  }
  int n = n0 + l15;
  if (n < N){
    #pragma unroll
    for (int r = 0; r < 8; ++r){
      int m = m0 + r + (hi ? 8 : 0);
      Y[yb + (size_t)m * N + n] = acc[r] + bias[m];
    }
  }
}

// -------- residual fuse: h += gelu(xw + xs) --------
__global__ __launch_bounds__(256) void fuse_kernel(
    float* __restrict__ h, const float* __restrict__ xw,
    const float* __restrict__ xs, size_t total){
  size_t i = (size_t)blockIdx.x * blockDim.x + threadIdx.x;
  if (i < total) h[i] += gelu_f(xw[i] + xs[i]);
}

// -------- host helpers --------
static inline void launch_pack(const float* X, _Float16* XT, int K, int N, int B,
                               hipStream_t s){
  dim3 g((N + 31) / 32, K / 32, B);
  packT_kernel<<<g, 256, 0, s>>>(X, XT, K, N);
}

static inline void launch_gemm(int Kv, int M, int N, int B, hipStream_t s,
                               const float* W, const float* bias, const _Float16* XT,
                               const float* Res, float* Y, int do_gelu){
  dim3 g((N + 127) / 128, M / 16, B);
  switch (Kv){
    case 32:  gemm_wmma_kernel<32> <<<g,128,0,s>>>(W,bias,XT,Res,Y,M,Kv,N,do_gelu); return;
    case 64:  gemm_wmma_kernel<64> <<<g,128,0,s>>>(W,bias,XT,Res,Y,M,Kv,N,do_gelu); return;
    case 128: gemm_wmma_kernel<128><<<g,128,0,s>>>(W,bias,XT,Res,Y,M,Kv,N,do_gelu); return;
    default:  gemm_wmma_kernel<0>  <<<g,128,0,s>>>(W,bias,XT,Res,Y,M,Kv,N,do_gelu); return;
  }
}

extern "C" void kernel_launch(void* const* d_in, const int* in_sizes, int n_in,
                              void* d_out, int out_size, void* d_ws, size_t ws_size,
                              hipStream_t stream){
  const float* x    = (const float*)d_in[0];
  const float* lw   = (const float*)d_in[1];
  const float* lb   = (const float*)d_in[2];
  const float* bww  = (const float*)d_in[3];
  const float* bwb  = (const float*)d_in[4];
  const float* bcaw = (const float*)d_in[5];
  const float* bcab = (const float*)d_in[6];
  const float* bcdw = (const float*)d_in[7];
  const float* bcdb = (const float*)d_in[8];
  const float* bm1w = (const float*)d_in[9];
  const float* bm1b = (const float*)d_in[10];
  const float* bm2w = (const float*)d_in[11];
  const float* bm2b = (const float*)d_in[12];
  const float* p1w  = (const float*)d_in[13];
  const float* p1b  = (const float*)d_in[14];
  const float* p2w  = (const float*)d_in[15];
  const float* p2b  = (const float*)d_in[16];

  const int S  = 8192;
  const int C  = in_sizes[1] / 2;          // 64
  const int B  = in_sizes[0] / S;          // 16
  const int NB = in_sizes[4] / C;          // 4
  const int LV = in_sizes[8] / (NB * C);   // 4
  const int Hh = in_sizes[10] / NB;        // 128
  const int Ch = in_sizes[14];             // 32
  const int NROW = B * C;

  int Ld[8]; Ld[0] = S;
  for (int i = 1; i <= LV; ++i) Ld[i] = (Ld[i-1] + 13) / 2 + 1;   // 4103,2059,1037,526
  size_t off[8]; off[1] = 0;
  for (int i = 2; i <= LV; ++i) off[i] = off[i-1] + (size_t)NROW * Ld[i-1];
  size_t WVN = off[LV] + (size_t)NROW * Ld[LV];

  float* ws = (float*)d_ws;
  const size_t HS = (size_t)NROW * S;
  float* H   = ws;                  // activation (persistent)
  float* XN  = ws + 1*HS;           // norm output / P1 scratch
  float* XS  = ws + 2*HS;           // conv3 output
  float* R0  = ws + 3*HS;           // waverec ping
  float* R1  = ws + 4*HS;           // waverec pong
  float* H2  = R0;                  // MLP hidden overlays R0..R1
  float* WA  = ws + 5*HS;           // dec approx chain
  float* WD  = WA + WVN;            // dec detail chain
  float* WMD = WD + WVN;            // mixed details
  float* WMA = WMD + WVN;           // mixed coarsest approx
  _Float16* PKT = (_Float16*)(WMA + (size_t)NROW * Ld[LV]);  // packed f16 scratch (B*N*K max)

  lift_kernel<<<dim3((unsigned)((HS + 255) / 256)), 256, 0, stream>>>(x, lw, lb, H, B, C, S);

  for (int blk = 0; blk < NB; ++blk){
    const float* ww  = bww  + (size_t)blk * C * C * 3;
    const float* wb  = bwb  + (size_t)blk * C;
    const float* caw = bcaw + (size_t)blk * C * C;
    const float* cab = bcab + (size_t)blk * C;
    const float* cdw = bcdw + (size_t)blk * LV * C * C;
    const float* cdb = bcdb + (size_t)blk * LV * C;
    const float* m1w = bm1w + (size_t)blk * Hh * C;
    const float* m1b = bm1b + (size_t)blk * Hh;
    const float* m2w = bm2w + (size_t)blk * C * Hh;
    const float* m2b = bm2b + (size_t)blk * C;

    inorm_kernel<<<NROW, 256, 0, stream>>>(H, XN, S);
    // conv3 on packed xn
    launch_pack(XN, PKT, C, S, B, stream);
    if (C == 64)
      conv3_wmma_kernel<64><<<dim3((S + 63) / 64, C / 16, B), 128, 0, stream>>>(ww, wb, PKT, XS, C, C, S);
    else
      conv3_wmma_kernel<0><<<dim3((S + 63) / 64, C / 16, B), 128, 0, stream>>>(ww, wb, PKT, XS, C, C, S);

    // analysis chain (f32)
    {
      const float* ain = XN; int Lin = S;
      for (int lv = 1; lv <= LV; ++lv){
        dwt_kernel<<<dim3((Ld[lv] + 255) / 256, NROW), 256, 0, stream>>>(
            ain, WA + off[lv], WD + off[lv], Lin, Ld[lv]);
        ain = WA + off[lv]; Lin = Ld[lv];
      }
    }
    // per-band channel mixes: pack band -> gemm (scratch reused sequentially)
    launch_pack(WA + off[LV], PKT, C, Ld[LV], B, stream);
    launch_gemm(C, C, Ld[LV], B, stream, caw, cab, PKT, nullptr, WMA, 0);
    for (int j = 0; j < LV; ++j){
      int lv = LV - j;
      launch_pack(WD + off[lv], PKT, C, Ld[lv], B, stream);
      launch_gemm(C, C, Ld[lv], B, stream, cdw + (size_t)j * C * C, cdb + (size_t)j * C,
                  PKT, nullptr, WMD + off[lv], 0);
    }
    // synthesis chain with trim
    {
      const float* abuf = WMA; int astride = Ld[LV]; int alen = Ld[LV];
      float* outb = R0; float* othb = R1;
      for (int lv = LV; lv >= 1; --lv){
        int dl = Ld[lv];
        int la = (alen == dl + 1) ? dl : alen;
        int lo = 2 * la - 14;
        idwt_kernel<<<dim3((lo + 255) / 256, NROW), 256, 0, stream>>>(
            abuf, WMD + off[lv], outb, la, lo, astride, dl);
        abuf = outb; astride = lo; alen = lo;
        float* t = outb; outb = othb; othb = t;
      }
      fuse_kernel<<<dim3((unsigned)((HS + 255) / 256)), 256, 0, stream>>>(H, abuf, XS, HS);
    }
    // MLP
    inorm_kernel<<<NROW, 256, 0, stream>>>(H, XN, S);
    launch_pack(XN, PKT, C, S, B, stream);
    launch_gemm(C, Hh, S, B, stream, m1w, m1b, PKT, nullptr, H2, 1);
    launch_pack(H2, PKT, Hh, S, B, stream);
    launch_gemm(Hh, C, S, B, stream, m2w, m2b, PKT, H, H, 0);
  }

  // head
  launch_pack(H, PKT, C, S, B, stream);
  launch_gemm(C, Ch, S, B, stream, p1w, p1b, PKT, nullptr, XN, 1);
  launch_pack(XN, PKT, Ch, S, B, stream);
  launch_gemm(Ch, C, S, B, stream, p2w, p2b, PKT, nullptr, (float*)d_out, 0);
}
